// GNN_Layer_32341103739523
// MI455X (gfx1250) — matrix operand order repaired and checked by reading, not compiled, not run on hardware
//
#include <hip/hip_runtime.h>
#include <cstdint>
#include <cstddef>

// ---------------------------------------------------------------------------
// GNN layer for MI455X (gfx1250): bf16 WMMA GEMMs + f32 atomic scatter.
// ---------------------------------------------------------------------------

#define N_NODES 40000
#define N_EDGES 640000
#define DIM_IN 128
#define DIM_EDGE 32
#define DIM_M 128
#define DIM_U 128
#define DIM_OUT 128

typedef __attribute__((ext_vector_type(16))) __bf16 v16bf;
typedef __attribute__((ext_vector_type(8)))  float  v8f;

union ABu { v16bf v; uint4 u[2]; };

__device__ __forceinline__ unsigned short f2bf(float f) {
  unsigned u = __float_as_uint(f);
  unsigned r = u + 0x7FFFu + ((u >> 16) & 1u);   // round-to-nearest-even
  return (unsigned short)(r >> 16);
}
__device__ __forceinline__ unsigned pk2(float a, float b) {
  return (unsigned)f2bf(a) | ((unsigned)f2bf(b) << 16);
}

// ---------------------------- prep kernels ---------------------------------

__global__ void cvt_f32_to_bf16(const float* __restrict__ src,
                                unsigned short* __restrict__ dst, int n) {
  int i = blockIdx.x * blockDim.x + threadIdx.x;
  if (i < n) dst[i] = f2bf(src[i]);
}

__global__ void zero_f32(float* __restrict__ p, int n) {
  int i = blockIdx.x * blockDim.x + threadIdx.x;
  if (i < n) p[i] = 0.0f;
}

// W is [K][N] row-major fp32 -> Wt is [N][K] bf16 (column of W contiguous).
__global__ void transpose_w(const float* __restrict__ w,
                            unsigned short* __restrict__ wt, int K, int N) {
  int i = blockIdx.x * blockDim.x + threadIdx.x;
  if (i < K * N) {
    int n = i / K, k = i % K;
    wt[n * K + k] = f2bf(w[(size_t)k * N + n]);
  }
}

// ----------------------------- edge kernel ---------------------------------
// 8 waves/block, one 16-edge tile per wave. 5000 blocks exactly cover E.
__global__ void edge_kernel(const float* __restrict__ ef,
                            const int* __restrict__ esrc,
                            const int* __restrict__ etgt,
                            const unsigned short* __restrict__ xbf,
                            const unsigned short* __restrict__ wm0t, // [128][160]
                            const unsigned short* __restrict__ wm1t, // [128][128]
                            const float* __restrict__ bM0,
                            const float* __restrict__ bM1,
                            float* __restrict__ agg) {
  extern __shared__ __align__(16) char smem[];
  unsigned short* sWM0 = (unsigned short*)smem;            // 20480 halfs
  unsigned short* sWM1 = (unsigned short*)(smem + 40960);  // 16384 halfs
  const int tid  = threadIdx.x;
  const int lane = tid & 31;
  const int wave = tid >> 5;
  unsigned short* sY = (unsigned short*)(smem + 73728) + wave * (16 * 128);

  { // cooperative weight staging: 4608 uint4 over 256 threads
    const uint4* g0 = (const uint4*)wm0t; uint4* s0 = (uint4*)sWM0;
    for (int i = tid; i < 2560; i += 256) s0[i] = g0[i];
    const uint4* g1 = (const uint4*)wm1t; uint4* s1 = (uint4*)sWM1;
    for (int i = tid; i < 2048; i += 256) s1[i] = g1[i];
  }
  __syncthreads();

  const int tile = blockIdx.x * 8 + wave;   // 0..39999
  const int e0   = tile * 16;
  const int half = lane >> 4;
  const int c    = lane & 15;

  const int myRow = e0 + c;                 // A-matrix row held by this lane
  const int src   = esrc[myRow];
  const unsigned short* xrow = xbf + (size_t)src * DIM_IN;

  // ---- layer 1: [16,160] x [160,128], K-tiles 0..4 ----
  v8f acc[8];
#pragma unroll
  for (int nt = 0; nt < 8; ++nt) {
    float b = bM0[nt * 16 + c];
#pragma unroll
    for (int j = 0; j < 8; ++j) acc[nt][j] = b;
  }

#pragma unroll
  for (int kt = 0; kt < 5; ++kt) {
    ABu a;
    if (kt < 4) {                           // from bf16 x
      int ks = kt * 32 + half * 8;
      a.u[0] = *(const uint4*)(xrow + ks);
      a.u[1] = *(const uint4*)(xrow + ks + 16);
    } else {                                // from fp32 edge features, inline cvt
      const float* efr = ef + (size_t)myRow * DIM_EDGE;
      int c0 = half * 8;
      float4 f0 = *(const float4*)(efr + c0);
      float4 f1 = *(const float4*)(efr + c0 + 4);
      float4 f2 = *(const float4*)(efr + c0 + 16);
      float4 f3 = *(const float4*)(efr + c0 + 20);
      a.u[0] = make_uint4(pk2(f0.x, f0.y), pk2(f0.z, f0.w),
                          pk2(f1.x, f1.y), pk2(f1.z, f1.w));
      a.u[1] = make_uint4(pk2(f2.x, f2.y), pk2(f2.z, f2.w),
                          pk2(f3.x, f3.y), pk2(f3.z, f3.w));
    }
#pragma unroll
    for (int nt = 0; nt < 8; ++nt) {
      ABu b;
      const uint4* bp =
          (const uint4*)(sWM0 + (nt * 16 + c) * 160 + kt * 32 + half * 8);
      b.u[0] = bp[0]; b.u[1] = bp[2];
      acc[nt] = __builtin_amdgcn_wmma_f32_16x16x32_bf16(
          false, a.v, false, b.v, (short)0, acc[nt], false, false);
    }
  }

  // ---- ReLU + restage y1 through LDS (C-layout -> row-major bf16) ----
#pragma unroll
  for (int nt = 0; nt < 8; ++nt)
#pragma unroll
    for (int j = 0; j < 8; ++j) {
      float v = acc[nt][j]; v = v > 0.0f ? v : 0.0f;
      sY[(j + 8 * half) * 128 + nt * 16 + c] = f2bf(v);
    }
  __syncthreads();

  // ---- layer 2: [16,128] x [128,128] ----
  v8f acc2[8];
#pragma unroll
  for (int nt = 0; nt < 8; ++nt) {
    float b = bM1[nt * 16 + c];
#pragma unroll
    for (int j = 0; j < 8; ++j) acc2[nt][j] = b;
  }
#pragma unroll
  for (int kt = 0; kt < 4; ++kt) {
    ABu a;
    const uint4* ap = (const uint4*)(sY + c * 128 + kt * 32 + half * 8);
    a.u[0] = ap[0];
    a.u[1] = *(const uint4*)(sY + c * 128 + kt * 32 + half * 8 + 16);
#pragma unroll
    for (int nt = 0; nt < 8; ++nt) {
      ABu b;
      const uint4* bp =
          (const uint4*)(sWM1 + (nt * 16 + c) * 128 + kt * 32 + half * 8);
      b.u[0] = bp[0]; b.u[1] = bp[2];
      acc2[nt] = __builtin_amdgcn_wmma_f32_16x16x32_bf16(
          false, a.v, false, b.v, (short)0, acc2[nt], false, false);
    }
  }

  // ---- ReLU + atomic scatter-add into agg[tgt] ----
#pragma unroll
  for (int j = 0; j < 8; ++j) {
    int m  = j + 8 * half;
    int tg = etgt[e0 + m];
    float* dst = agg + (size_t)tg * DIM_M + c;
#pragma unroll
    for (int nt = 0; nt < 8; ++nt) {
      float v = acc2[nt][j]; v = v > 0.0f ? v : 0.0f;
      atomicAdd(dst + nt * 16, v);
    }
  }
}

// ----------------------------- node kernel ---------------------------------
// 4 waves/block, one 16-node tile per wave. 625 blocks exactly cover N.
__global__ void node_kernel(const unsigned short* __restrict__ xbf,
                            const float* __restrict__ agg,
                            const unsigned short* __restrict__ wu0t, // [128][256]
                            const unsigned short* __restrict__ wuot, // [128][128]
                            const float* __restrict__ bU0,
                            const float* __restrict__ bUo,
                            float* __restrict__ out) {
  extern __shared__ __align__(16) char smem[];
  unsigned short* sWU0 = (unsigned short*)smem;            // 32768 halfs
  unsigned short* sWUo = (unsigned short*)(smem + 65536);  // 16384 halfs
  const int tid  = threadIdx.x;
  const int lane = tid & 31;
  const int wave = tid >> 5;
  unsigned short* sZ = (unsigned short*)(smem + 98304) + wave * (16 * 128);

  { // 6144 uint4 over 128 threads
    const uint4* g0 = (const uint4*)wu0t; uint4* s0 = (uint4*)sWU0;
    for (int i = tid; i < 4096; i += 128) s0[i] = g0[i];
    const uint4* g1 = (const uint4*)wuot; uint4* s1 = (uint4*)sWUo;
    for (int i = tid; i < 2048; i += 128) s1[i] = g1[i];
  }
  __syncthreads();

  const int tile = blockIdx.x * 4 + wave;  // 0..2499
  const int n0   = tile * 16;
  const int half = lane >> 4;
  const int c    = lane & 15;

  const int myRow = n0 + c;
  const unsigned short* xrow = xbf + (size_t)myRow * DIM_IN;
  const float* arow          = agg + (size_t)myRow * DIM_M;

  // ---- layer 1: [16,256] x [256,128], K-tiles 0..7 ----
  v8f acc[8];
#pragma unroll
  for (int nt = 0; nt < 8; ++nt) {
    float b = bU0[nt * 16 + c];
#pragma unroll
    for (int j = 0; j < 8; ++j) acc[nt][j] = b;
  }
#pragma unroll
  for (int kt = 0; kt < 8; ++kt) {
    ABu a;
    if (kt < 4) {                           // x half of concat (bf16)
      int ks = kt * 32 + half * 8;
      a.u[0] = *(const uint4*)(xrow + ks);
      a.u[1] = *(const uint4*)(xrow + ks + 16);
    } else {                                // agg half of concat (fp32 -> bf16)
      int ks = (kt - 4) * 32 + half * 8;
      float4 f0 = *(const float4*)(arow + ks);
      float4 f1 = *(const float4*)(arow + ks + 4);
      float4 f2 = *(const float4*)(arow + ks + 16);
      float4 f3 = *(const float4*)(arow + ks + 20);
      a.u[0] = make_uint4(pk2(f0.x, f0.y), pk2(f0.z, f0.w),
                          pk2(f1.x, f1.y), pk2(f1.z, f1.w));
      a.u[1] = make_uint4(pk2(f2.x, f2.y), pk2(f2.z, f2.w),
                          pk2(f3.x, f3.y), pk2(f3.z, f3.w));
    }
#pragma unroll
    for (int nt = 0; nt < 8; ++nt) {
      ABu b;
      const uint4* bp =
          (const uint4*)(sWU0 + (nt * 16 + c) * 256 + kt * 32 + half * 8);
      b.u[0] = bp[0]; b.u[1] = bp[2];
      acc[nt] = __builtin_amdgcn_wmma_f32_16x16x32_bf16(
          false, a.v, false, b.v, (short)0, acc[nt], false, false);
    }
  }

  // ---- ReLU + restage z1 through LDS ----
#pragma unroll
  for (int nt = 0; nt < 8; ++nt)
#pragma unroll
    for (int j = 0; j < 8; ++j) {
      float v = acc[nt][j]; v = v > 0.0f ? v : 0.0f;
      sZ[(j + 8 * half) * 128 + nt * 16 + c] = f2bf(v);
    }
  __syncthreads();

  // ---- output layer: [16,128] x [128,128], no activation ----
  v8f acc2[8];
#pragma unroll
  for (int nt = 0; nt < 8; ++nt) {
    float b = bUo[nt * 16 + c];
#pragma unroll
    for (int j = 0; j < 8; ++j) acc2[nt][j] = b;
  }
#pragma unroll
  for (int kt = 0; kt < 4; ++kt) {
    ABu a;
    const uint4* ap = (const uint4*)(sZ + c * 128 + kt * 32 + half * 8);
    a.u[0] = ap[0];
    a.u[1] = *(const uint4*)(sZ + c * 128 + kt * 32 + half * 8 + 16);
#pragma unroll
    for (int nt = 0; nt < 8; ++nt) {
      ABu b;
      const uint4* bp =
          (const uint4*)(sWUo + (nt * 16 + c) * 128 + kt * 32 + half * 8);
      b.u[0] = bp[0]; b.u[1] = bp[2];
      acc2[nt] = __builtin_amdgcn_wmma_f32_16x16x32_bf16(
          false, a.v, false, b.v, (short)0, acc2[nt], false, false);
    }
  }

#pragma unroll
  for (int j = 0; j < 8; ++j) {
    int node = n0 + j + 8 * half;
#pragma unroll
    for (int nt = 0; nt < 8; ++nt)
      out[(size_t)node * DIM_OUT + nt * 16 + c] = acc2[nt][j];
  }
}

// ------------------------------- launch ------------------------------------

extern "C" void kernel_launch(void* const* d_in, const int* in_sizes, int n_in,
                              void* d_out, int out_size, void* d_ws, size_t ws_size,
                              hipStream_t stream) {
  const float* x    = (const float*)d_in[0];
  const float* ef   = (const float*)d_in[1];
  const int*   eidx = (const int*)d_in[2];
  // d_in[3] = batch_idx (unused by the reference computation)
  const float* WM0 = (const float*)d_in[4];
  const float* bM0 = (const float*)d_in[5];
  const float* WM1 = (const float*)d_in[6];
  const float* bM1 = (const float*)d_in[7];
  const float* WU0 = (const float*)d_in[8];
  const float* bU0 = (const float*)d_in[9];
  const float* WUo = (const float*)d_in[10];
  const float* bUo = (const float*)d_in[11];
  float* out = (float*)d_out;

  const int* esrc = eidx;
  const int* etgt = eidx + N_EDGES;

  // workspace layout (bytes)
  char* ws = (char*)d_ws;
  unsigned short* xbf  = (unsigned short*)(ws);                 // 10,240,000 B
  float*          agg  = (float*)(ws + 10240000);               // 20,480,000 B
  unsigned short* wm0t = (unsigned short*)(ws + 30720000);      //     40,960 B
  unsigned short* wm1t = (unsigned short*)(ws + 30760960);      //     32,768 B
  unsigned short* wu0t = (unsigned short*)(ws + 30793728);      //     65,536 B
  unsigned short* wuot = (unsigned short*)(ws + 30859264);      //     32,768 B

  // ---- prep: bf16 x, transposed bf16 weights, zeroed agg ----
  {
    int n = N_NODES * DIM_IN;  // 5,120,000
    cvt_f32_to_bf16<<<(n + 255) / 256, 256, 0, stream>>>(x, xbf, n);
    zero_f32<<<(n + 255) / 256, 256, 0, stream>>>(agg, n);
  }
  transpose_w<<<(160 * 128 + 255) / 256, 256, 0, stream>>>(WM0, wm0t, 160, 128);
  transpose_w<<<(128 * 128 + 255) / 256, 256, 0, stream>>>(WM1, wm1t, 128, 128);
  transpose_w<<<(256 * 128 + 255) / 256, 256, 0, stream>>>(WU0, wu0t, 256, 128);
  transpose_w<<<(128 * 128 + 255) / 256, 256, 0, stream>>>(WUo, wuot, 128, 128);

  // ---- edge MLP + scatter: 40000 tiles, 8 waves/block ----
  {
    size_t lds = 40960 + 32768 + 8 * (16 * 128 * 2);  // 106,496 B (WGP has 320 KB)
    edge_kernel<<<N_EDGES / (16 * 8), 256, lds, stream>>>(
        ef, esrc, etgt, xbf, wm0t, wm1t, bM0, bM1, agg);
  }

  // ---- node MLP: 2500 tiles, 4 waves/block ----
  {
    size_t lds = 65536 + 32768 + 4 * (16 * 128 * 2);  // 114,688 B
    node_kernel<<<N_NODES / (16 * 4), 128, lds, stream>>>(
        xbf, agg, wu0t, wuot, bU0, bUo, out);
  }
}
